// Model_Net_19516331393570
// MI455X (gfx1250) — matrix operand order repaired and checked by reading, not compiled, not run on hardware
//
#include <hip/hip_runtime.h>
#include <math.h>

#define NNODES 30000
#define NEDGES 480000
#define NGRAPHS 64
#define AAC 21
#define HID 128
#define DIN 6165
#define C1D 149
#define C2D 298
#define C3D 596
#define FC1D 1024
#define OUTD 486

#define XROWS 30080    // 235*128 : every row-tile fully allocated
#define XLDH  640      // f16 node-feature / pooled row stride (mult of 64)
#define HLD   596      // fp32 h / agg row stride

typedef __attribute__((ext_vector_type(16))) _Float16 v16h;
typedef __attribute__((ext_vector_type(8)))  _Float16 v8h;
typedef __attribute__((ext_vector_type(8)))  float    v8f;

// ---- CDNA5 async global->LDS path (guarded; sync fallback keeps compile green)
#if defined(__HIP_DEVICE_COMPILE__)
#if __has_builtin(__builtin_amdgcn_global_load_async_to_lds_b128) && \
    __has_builtin(__builtin_amdgcn_s_wait_asynccnt)
#define HAVE_ASYNC 1
#endif
#endif
#ifndef HAVE_ASYNC
#define HAVE_ASYNC 0
#endif

#if HAVE_ASYNC
typedef int v4i __attribute__((vector_size(4 * sizeof(int))));
typedef __attribute__((address_space(1))) v4i GV4;   // global int4*
typedef __attribute__((address_space(3))) v4i LV4;   // LDS int4*
#define ASYNC_CP16(gp, lp)                                              \
    __builtin_amdgcn_global_load_async_to_lds_b128(                     \
        (GV4*)(unsigned long long)(gp), (LV4*)(lp), 0, 0)
#define WAIT_ASYNC(n) __builtin_amdgcn_s_wait_asynccnt(n)
#endif

// ---------------------------------------------------------------------------
// shared WMMA helpers
// ---------------------------------------------------------------------------
__device__ __forceinline__ void mma_chunk(const _Float16 (*As)[72], const _Float16 (*Bs)[72],
                                          int wave, int idx16, int halfsel, v8f* acc)
{
    #pragma unroll
    for (int kk = 0; kk < 64; kk += 32) {
        // A frag: elems 0..7 -> K = kk+8*half+e ; elems 8..15 -> +16
        const _Float16* ap = &As[wave * 16 + idx16][kk + 8 * halfsel];
        v8h a0 = *(const v8h*)ap, a1 = *(const v8h*)(ap + 16);
        v16h a;
        #pragma unroll
        for (int e = 0; e < 8; ++e) { a[e] = a0[e]; a[8 + e] = a1[e]; }
        #pragma unroll
        for (int nt = 0; nt < 8; ++nt) {
            // B frag: elems 0..7 -> K = kk+8*half+e ; elems 8..15 -> +16
            const _Float16* bp = &Bs[nt * 16 + idx16][kk + 8 * halfsel];
            v8h b0 = *(const v8h*)bp, b1 = *(const v8h*)(bp + 16);
            v16h b;
            #pragma unroll
            for (int e = 0; e < 8; ++e) { b[e] = b0[e]; b[8 + e] = b1[e]; }
            acc[nt] = __builtin_amdgcn_wmma_f32_16x16x32_f16(
                false, a, false, b, (short)0, acc[nt], false, false);
        }
    }
}

__device__ __forceinline__ void gemm_epilogue(v8f* acc, const float* bias, void* Cp,
    int ldc, int col_off, int c_f16, int M, int N, int mode,
    long long rowBase, int colBase, int wave, int idx16, int halfsel)
{
    #pragma unroll
    for (int nt = 0; nt < 8; ++nt) {
        int gn = colBase + nt * 16 + idx16;
        if (gn >= N) continue;
        float bv = bias ? bias[gn] : 0.0f;
        #pragma unroll
        for (int r = 0; r < 8; ++r) {
            long long gm = rowBase + wave * 16 + r + 8 * halfsel;
            if (gm >= M) continue;
            float v = acc[nt][r] + bv;
            if (mode == 1)      v = fmaxf(v, 0.0f);
            else if (mode == 2) v = 1.0f / (1.0f + __expf(-v));
            long long idx = gm * (long long)ldc + col_off + gn;
            if (c_f16) ((_Float16*)Cp)[idx] = (_Float16)v;
            else       ((float*)Cp)[idx] = v;
        }
    }
}

// ---------------------------------------------------------------------------
// Tiled WMMA GEMM. AF16: A is padded f16 (rows fully allocated, K zero-padded)
//   -> async double-buffered A staging. Otherwise A is fp32 with bounds+convert.
// B is always pre-converted, pre-transposed f16: Bt[Npad][Kpad].
// mode: 0 none, 1 relu, 2 sigmoid. c_f16: output element type.
// ---------------------------------------------------------------------------
template<bool AF16>
__global__ __launch_bounds__(256)
void wmma_gemm(const void* __restrict__ Ap, int lda,
               const _Float16* __restrict__ Bt, int ldbt,
               const float* __restrict__ bias,
               void* __restrict__ Cp, int ldc, int col_off, int c_f16,
               int M, int N, int K, int mode)
{
    __shared__ __align__(16) _Float16 As[2][128][72];  // A double buffer
    __shared__ __align__(16) _Float16 Bs[128][72];     // Bs[n][k]

    const int tid  = threadIdx.x;
    const int lane = tid & 31;
    const int wave = tid >> 5;
    const long long rowBase = (long long)blockIdx.x * 128;
    const int colBase = blockIdx.y * 128;
    const int idx16 = lane & 15, halfsel = lane >> 4;
    const int nch = (K + 63) >> 6;

    v8f acc[8];
    v8f z = {};
    #pragma unroll
    for (int nt = 0; nt < 8; ++nt) acc[nt] = z;

#if HAVE_ASYNC
    if constexpr (AF16) {
        const _Float16* A = (const _Float16*)Ap;
        // prologue: async copy chunk 0 A tile (4 b128 instrs per wave)
        #pragma unroll
        for (int j = 0; j < 4; ++j) {
            int flat = j * 256 + tid, r = flat >> 3, s = flat & 7;
            ASYNC_CP16(A + (rowBase + r) * (long long)lda + s * 8, &As[0][r][s * 8]);
        }
        for (int c = 0; c < nch; ++c) {
            const int cur = c & 1, k0 = c << 6;
            __syncthreads();                      // prev readers done with As[1-cur] & Bs
            if (c + 1 < nch) {
                #pragma unroll
                for (int j = 0; j < 4; ++j) {
                    int flat = j * 256 + tid, r = flat >> 3, s = flat & 7;
                    ASYNC_CP16(A + (rowBase + r) * (long long)lda + k0 + 64 + s * 8,
                               &As[1 - cur][r][s * 8]);
                }
            }
            // stage B chunk c (weights: L2-resident, sync copy)
            #pragma unroll
            for (int j = 0; j < 4; ++j) {
                int flat = j * 256 + tid, r = flat >> 3, s = flat & 7;
                *(v8h*)&Bs[r][s * 8] =
                    *(const v8h*)(Bt + (long long)(colBase + r) * ldbt + k0 + s * 8);
            }
            if (c + 1 < nch) WAIT_ASYNC(4); else WAIT_ASYNC(0);
            __syncthreads();                      // all waves' A async + B stores visible
            mma_chunk(As[cur], Bs, wave, idx16, halfsel, acc);
        }
        gemm_epilogue(acc, bias, Cp, ldc, col_off, c_f16, M, N, mode,
                      rowBase, colBase, wave, idx16, halfsel);
        return;
    }
#endif
    // ---- synchronous path (fp32-A convert staging, or f16 fallback) ----
    for (int c = 0; c < nch; ++c) {
        int k0 = c << 6;
        if constexpr (AF16) {
            const _Float16* A = (const _Float16*)Ap;
            #pragma unroll
            for (int j = 0; j < 4; ++j) {
                int flat = j * 256 + tid, r = flat >> 3, s = flat & 7;
                *(v8h*)&As[0][r][s * 8] =
                    *(const v8h*)(A + (rowBase + r) * (long long)lda + k0 + s * 8);
            }
        } else {
            const float* A = (const float*)Ap;
            for (int i = tid; i < 128 * 64; i += 256) {
                int r = i >> 6, cc = i & 63;
                long long gr = rowBase + r;
                int gc = k0 + cc;
                float v = (gr < M && gc < K) ? A[gr * (long long)lda + gc] : 0.0f;
                As[0][r][cc] = (_Float16)v;
            }
            if (k0 + 64 < K) {   // global_prefetch_b8 for next A chunk
                long long pr = rowBase + (tid >> 1);
                if (pr < M)
                    __builtin_prefetch(A + pr * (long long)lda + k0 + 64 + ((tid & 1) << 5), 0, 1);
            }
        }
        #pragma unroll
        for (int j = 0; j < 4; ++j) {
            int flat = j * 256 + tid, r = flat >> 3, s = flat & 7;
            *(v8h*)&Bs[r][s * 8] =
                *(const v8h*)(Bt + (long long)(colBase + r) * ldbt + k0 + s * 8);
        }
        __syncthreads();
        mma_chunk(As[0], Bs, wave, idx16, halfsel, acc);
        __syncthreads();
    }
    gemm_epilogue(acc, bias, Cp, ldc, col_off, c_f16, M, N, mode,
                  rowBase, colBase, wave, idx16, halfsel);
}

// ---------------------------------------------------------------------------
// Helper kernels
// ---------------------------------------------------------------------------
__global__ void fill_f32(float* p, long long n, float v) {
    long long i = (long long)blockIdx.x * blockDim.x + threadIdx.x;
    long long st = (long long)gridDim.x * blockDim.x;
    for (; i < n; i += st) p[i] = v;
}

__global__ void fill_f16(_Float16* p, long long n, float v) {
    long long i = (long long)blockIdx.x * blockDim.x + threadIdx.x;
    long long st = (long long)gridDim.x * blockDim.x;
    _Float16 h = (_Float16)v;
    for (; i < n; i += st) p[i] = h;
}

// Wt[n*Kpad + k] = W[k*N + n] (f16, zero padded)
__global__ void wt_kernel(const float* __restrict__ W, int K, int N,
                          _Float16* __restrict__ Wt, int Kpad, int Npad) {
    int i = blockIdx.x * blockDim.x + threadIdx.x;
    if (i >= Kpad * Npad) return;
    int n = i / Kpad, k = i % Kpad;
    float v = (n < N && k < K) ? W[(long long)k * N + n] : 0.0f;
    Wt[i] = (_Float16)v;
}

// x[:, 0:21] = relu(prot_x[:, :21] @ W2 + b2)   (f16 out)
__global__ void f2_kernel(const float* __restrict__ px, const float* __restrict__ W2,
                          const float* __restrict__ b2, _Float16* __restrict__ x) {
    int i = blockIdx.x * blockDim.x + threadIdx.x;
    if (i >= NNODES * AAC) return;
    int node = i / AAC, j = i % AAC;
    float s = b2[j];
    #pragma unroll
    for (int k = 0; k < AAC; ++k)
        s += px[(long long)node * DIN + k] * W2[k * AAC + j];
    x[(long long)node * XLDH + j] = (_Float16)fmaxf(s, 0.0f);
}

__global__ void deg_kernel(const int* __restrict__ dst, float* __restrict__ deg) {
    int e = blockIdx.x * blockDim.x + threadIdx.x;
    if (e < NEDGES) atomicAdd(&deg[dst[e]], 1.0f);
}

__global__ void dinv_kernel(const float* __restrict__ deg, float* __restrict__ dinv,
                            float* __restrict__ invdeg) {
    int i = blockIdx.x * blockDim.x + threadIdx.x;
    if (i < NNODES) { float d = deg[i]; dinv[i] = rsqrtf(d); invdeg[i] = 1.0f / d; }
}

// one block per edge: agg[dst] += h[src] * dinv[src]*dinv[dst]
__global__ __launch_bounds__(128)
void scatter_kernel(const int* __restrict__ src, const int* __restrict__ dst,
                    const float* __restrict__ dinv, const float* __restrict__ h,
                    float* __restrict__ agg, int C) {
    int e = blockIdx.x;
    int s = src[e], d = dst[e];
    float w = dinv[s] * dinv[d];
    const float* hs = h + (long long)s * HLD;
    float* ad = agg + (long long)d * HLD;
    for (int c = threadIdx.x; c < C; c += blockDim.x)
        atomicAdd(&ad[c], hs[c] * w);
}

// x_out(f16) = relu(agg + invdeg * h + bias)
__global__ void combine_kernel(const float* __restrict__ agg, const float* __restrict__ h,
                               const float* __restrict__ invdeg, const float* __restrict__ bias,
                               _Float16* __restrict__ xout, int C) {
    long long i = (long long)blockIdx.x * blockDim.x + threadIdx.x;
    long long total = (long long)NNODES * C;
    if (i >= total) return;
    int node = (int)(i / C), c = (int)(i % C);
    float v = agg[(long long)node * HLD + c] + invdeg[node] * h[(long long)node * HLD + c] + bias[c];
    xout[(long long)node * XLDH + c] = (_Float16)fmaxf(v, 0.0f);
}

__global__ __launch_bounds__(128)
void pool_sum_kernel(const _Float16* __restrict__ x, const int* __restrict__ batch,
                     float* __restrict__ sums, float* __restrict__ cnts) {
    int node = blockIdx.x;
    int g = batch[node];
    if (threadIdx.x == 0) atomicAdd(&cnts[g], 1.0f);
    const _Float16* xr = x + (long long)node * XLDH;
    float* sg = sums + (long long)g * C3D;
    for (int c = threadIdx.x; c < C3D; c += blockDim.x)
        atomicAdd(&sg[c], (float)xr[c]);
}

__global__ void pool_div_kernel(const float* __restrict__ sums, const float* __restrict__ cnts,
                                _Float16* __restrict__ pooled) {
    int i = blockIdx.x * blockDim.x + threadIdx.x;
    if (i >= NGRAPHS * C3D) return;
    int g = i / C3D, c = i % C3D;
    pooled[(long long)g * XLDH + c] = (_Float16)(sums[i] / fmaxf(cnts[g], 1.0f));
}

__global__ void bn_relu_kernel(_Float16* __restrict__ y, const float* __restrict__ gamma,
                               const float* __restrict__ beta) {
    int i = blockIdx.x * blockDim.x + threadIdx.x;
    if (i >= NGRAPHS * FC1D) return;
    int c = i % FC1D;
    const float invs = 0.99999500003749968f;  // 1/sqrt(1 + 1e-5)
    float v = (float)y[i] * invs * gamma[c] + beta[c];
    y[i] = (_Float16)fmaxf(v, 0.0f);
}

// ---------------------------------------------------------------------------
// Orchestration
// ---------------------------------------------------------------------------
static inline size_t alignUp(size_t x) { return (x + 255) & ~(size_t)255; }

extern "C" void kernel_launch(void* const* d_in, const int* in_sizes, int n_in,
                              void* d_out, int out_size, void* d_ws, size_t ws_size,
                              hipStream_t stream) {
    const float* prot_x = (const float*)d_in[0];
    const int*   src    = (const int*)d_in[1];
    const int*   dst    = (const int*)d_in[2];
    const int*   batch  = (const int*)d_in[3];
    const float* W1 = (const float*)d_in[4];   const float* b1 = (const float*)d_in[5];
    const float* W2 = (const float*)d_in[6];   const float* b2 = (const float*)d_in[7];
    const float* Wc[3] = { (const float*)d_in[8], (const float*)d_in[10], (const float*)d_in[12] };
    const float* bc[3] = { (const float*)d_in[9], (const float*)d_in[11], (const float*)d_in[13] };
    const float* Wg1 = (const float*)d_in[14]; const float* bg1 = (const float*)d_in[15];
    const float* Wg2 = (const float*)d_in[16]; const float* bg2 = (const float*)d_in[17];
    const float* gamma = (const float*)d_in[18];
    const float* beta  = (const float*)d_in[19];
    float* out = (float*)d_out;
    (void)in_sizes; (void)n_in; (void)out_size; (void)ws_size;

    // ---- workspace layout ----
    char* ws = (char*)d_ws;
    size_t off = 0;
    _Float16* bufX = (_Float16*)(ws + off); off = alignUp(off + (size_t)XROWS * XLDH * 2);
    float* bufH = (float*)(ws + off); off = alignUp(off + (size_t)NNODES * HLD * 4);
    float* bufA = (float*)(ws + off); off = alignUp(off + (size_t)NNODES * HLD * 4);
    float* deg    = (float*)(ws + off); off = alignUp(off + NNODES * 4);
    float* dinv   = (float*)(ws + off); off = alignUp(off + NNODES * 4);
    float* invdeg = (float*)(ws + off); off = alignUp(off + NNODES * 4);
    float* sums   = (float*)(ws + off); off = alignUp(off + (size_t)NGRAPHS * C3D * 4);
    float* cnts   = (float*)(ws + off); off = alignUp(off + NGRAPHS * 4);
    _Float16* pooled = (_Float16*)(ws + off); off = alignUp(off + (size_t)128 * XLDH * 2);
    _Float16* fc1    = (_Float16*)(ws + off); off = alignUp(off + (size_t)128 * FC1D * 2);
    // pre-converted transposed f16 weights [Npad][Kpad]
    _Float16* Wt1  = (_Float16*)(ws + off); off = alignUp(off + (size_t)128  * 6144 * 2);
    _Float16* Wtc1 = (_Float16*)(ws + off); off = alignUp(off + (size_t)256  * 192  * 2);
    _Float16* Wtc2 = (_Float16*)(ws + off); off = alignUp(off + (size_t)384  * 192  * 2);
    _Float16* Wtc3 = (_Float16*)(ws + off); off = alignUp(off + (size_t)640  * 320  * 2);
    _Float16* Wtg1 = (_Float16*)(ws + off); off = alignUp(off + (size_t)1024 * 640  * 2);
    _Float16* Wtg2 = (_Float16*)(ws + off); off = alignUp(off + (size_t)512  * 1024 * 2);

    const int mtiles = (NNODES + 127) / 128;   // 235

    // ---- weight conversion (f16, transposed, padded) ----
    wt_kernel<<<(128 * 6144 + 255) / 256, 256, 0, stream>>>(W1, 6144, HID, Wt1, 6144, 128);
    wt_kernel<<<(256 * 192 + 255) / 256, 256, 0, stream>>>(Wc[0], C1D, C1D, Wtc1, 192, 256);
    wt_kernel<<<(384 * 192 + 255) / 256, 256, 0, stream>>>(Wc[1], C1D, C2D, Wtc2, 192, 384);
    wt_kernel<<<(640 * 320 + 255) / 256, 256, 0, stream>>>(Wc[2], C2D, C3D, Wtc3, 320, 640);
    wt_kernel<<<(1024 * 640 + 255) / 256, 256, 0, stream>>>(Wg1, C3D, FC1D, Wtg1, 640, 1024);
    wt_kernel<<<(512 * 1024 + 255) / 256, 256, 0, stream>>>(Wg2, FC1D, OUTD, Wtg2, 1024, 512);

    // ---- zero-init padded f16 buffers (K-pad columns must be zero) ----
    fill_f16<<<2048, 256, 0, stream>>>(bufX, (long long)XROWS * XLDH, 0.0f);
    fill_f16<<<128, 256, 0, stream>>>(pooled, (long long)128 * XLDH, 0.0f);

    // ---- degrees (self-loop included: init 1) ----
    fill_f32<<<256, 256, 0, stream>>>(deg, NNODES, 1.0f);
    deg_kernel<<<(NEDGES + 255) / 256, 256, 0, stream>>>(dst, deg);
    dinv_kernel<<<(NNODES + 255) / 256, 256, 0, stream>>>(deg, dinv, invdeg);

    // ---- node embedding: x = [relu(px[:, :21]W2+b2), relu(px[:, 21:]W1+b1)] ----
    f2_kernel<<<(NNODES * AAC + 255) / 256, 256, 0, stream>>>(prot_x, W2, b2, bufX);
    wmma_gemm<false><<<dim3(mtiles, 1), 256, 0, stream>>>(
        prot_x + AAC, DIN, Wt1, 6144, b1, bufX, XLDH, AAC, 1,
        NNODES, HID, DIN - AAC, 1);

    // ---- three GCN convs ----
    const int cin[3]  = { C1D, C1D, C2D };
    const int cout[3] = { C1D, C2D, C3D };
    const _Float16* Wt[3] = { Wtc1, Wtc2, Wtc3 };
    const int kpad[3] = { 192, 192, 320 };
    for (int l = 0; l < 3; ++l) {
        wmma_gemm<true><<<dim3(mtiles, (cout[l] + 127) / 128), 256, 0, stream>>>(
            bufX, XLDH, Wt[l], kpad[l], nullptr, bufH, HLD, 0, 0,
            NNODES, cout[l], cin[l], 0);
        fill_f32<<<2048, 256, 0, stream>>>(bufA, (long long)NNODES * HLD, 0.0f);
        scatter_kernel<<<NEDGES, 128, 0, stream>>>(src, dst, dinv, bufH, bufA, cout[l]);
        long long tot = (long long)NNODES * cout[l];
        combine_kernel<<<(unsigned)((tot + 255) / 256), 256, 0, stream>>>(
            bufA, bufH, invdeg, bc[l], bufX, cout[l]);
    }

    // ---- global mean pool ----
    fill_f32<<<64, 256, 0, stream>>>(sums, (long long)NGRAPHS * C3D, 0.0f);
    fill_f32<<<1, 64, 0, stream>>>(cnts, NGRAPHS, 0.0f);
    pool_sum_kernel<<<NNODES, 128, 0, stream>>>(bufX, batch, sums, cnts);
    pool_div_kernel<<<(NGRAPHS * C3D + 255) / 256, 256, 0, stream>>>(sums, cnts, pooled);

    // ---- FC1 + BN + relu ----
    wmma_gemm<true><<<dim3(1, 8), 256, 0, stream>>>(
        pooled, XLDH, Wtg1, 640, bg1, fc1, FC1D, 0, 1,
        NGRAPHS, FC1D, C3D, 0);
    bn_relu_kernel<<<(NGRAPHS * FC1D + 255) / 256, 256, 0, stream>>>(fc1, gamma, beta);

    // ---- FC2 + sigmoid -> output [64, 486] fp32 ----
    wmma_gemm<true><<<dim3(1, 4), 256, 0, stream>>>(
        fc1, FC1D, Wtg2, 1024, bg2, out, OUTD, 0, 0,
        NGRAPHS, OUTD, FC1D, 2);
}
